// ScaledDotProductAttention_69844758167798
// MI455X (gfx1250) — matrix operand order repaired and checked by reading, not compiled, not run on hardware
//
#include <hip/hip_runtime.h>

#define BB 2
#define SS 2048
#define DDIM 512
#define HH 8
#define HD 64

// gate LDS tile: 32 k-rows, each 16q x 8h = 128 floats, padded to 132 for banks
#define GROW 132
#define GTILE (32 * GROW)

typedef __attribute__((ext_vector_type(16))) __bf16 v16bf;
typedef __attribute__((ext_vector_type(8)))  __bf16 v8bf;
typedef __attribute__((ext_vector_type(8)))  float  v8f;
typedef __attribute__((ext_vector_type(4)))  float  f32x4;
typedef int v4i __attribute__((vector_size(16)));

union U16 { v16bf v; v8bf h[2]; };

#if defined(__has_builtin)
#if __has_builtin(__builtin_amdgcn_global_load_async_to_lds_b128)
#define HAVE_ASYNC 1
#endif
#endif

__device__ __forceinline__ void lds_wait0() {
#if defined(__has_builtin) && __has_builtin(__builtin_amdgcn_s_wait_dscnt)
  __builtin_amdgcn_s_wait_dscnt(0);
  asm volatile("" ::: "memory");
#else
  asm volatile("s_wait_dscnt 0x0" ::: "memory");
#endif
}

#ifdef HAVE_ASYNC
__device__ __forceinline__ void async_wait0() {
#if __has_builtin(__builtin_amdgcn_s_wait_asynccnt)
  __builtin_amdgcn_s_wait_asynccnt(0);
  asm volatile("" ::: "memory");
#else
  asm volatile("s_wait_asynccnt 0x0" ::: "memory");
#endif
}
__device__ __forceinline__ void async_wait4() {
#if __has_builtin(__builtin_amdgcn_s_wait_asynccnt)
  __builtin_amdgcn_s_wait_asynccnt(4);
  asm volatile("" ::: "memory");
#else
  asm volatile("s_wait_asynccnt 0x4" ::: "memory");
#endif
}
// stage 64B (4 x b128) global -> LDS, async (per-lane addresses)
__device__ __forceinline__ void stage16(const float* gsrc, float* ldst) {
#pragma unroll
  for (int i = 0; i < 4; ++i) {
    __builtin_amdgcn_global_load_async_to_lds_b128(
        (v4i*)(gsrc + 4 * i), (v4i*)(ldst + 4 * i), 0, 0);
  }
}
#else
__device__ __forceinline__ void stage16_sync(const float* gsrc, float* ldst) {
#pragma unroll
  for (int i = 0; i < 4; ++i) {
    ((f32x4*)ldst)[i] = ((const f32x4*)gsrc)[i];
  }
}
#endif

__device__ __forceinline__ __bf16 f2bf(float f) {
  unsigned u = __builtin_bit_cast(unsigned, f);
  unsigned r = u + 0x7FFFu + ((u >> 16) & 1u);   // round-to-nearest-even
  unsigned short hs = (unsigned short)(r >> 16);
  return __builtin_bit_cast(__bf16, hs);
}

__global__ void cvt_f32_to_bf16(const float* __restrict__ in,
                                __bf16* __restrict__ out, int n) {
  int i = blockIdx.x * blockDim.x + threadIdx.x;
  if (i < n) out[i] = f2bf(in[i]);
}

// One wave computes one 16x16 tile of P = X * W^T + bias.
// vtrans=0: out[b][h][s][d] ; vtrans=1: out[b][h][d][s]
__global__ __launch_bounds__(32)
void proj16x16(const __bf16* __restrict__ X, const __bf16* __restrict__ W,
               const float* __restrict__ bias, __bf16* __restrict__ out,
               int vtrans) {
  const int l  = threadIdx.x;
  const int ln = l & 15;
  const int hl = l >> 4;
  const int m0 = blockIdx.x * 16;
  const int n0 = blockIdx.y * 16;

  v8f acc = {};
  const __bf16* xr = X + (size_t)(m0 + ln) * DDIM;
  const __bf16* wr = W + (size_t)(n0 + ln) * DDIM;
  for (int kb = 0; kb < DDIM; kb += 32) {
    U16 a, bfrag;
    a.h[0]  = *(const v8bf*)(xr + kb + 8 * hl);
    a.h[1]  = *(const v8bf*)(xr + kb + 16 + 8 * hl);
    bfrag.h[0] = *(const v8bf*)(wr + kb + 8 * hl);
    bfrag.h[1] = *(const v8bf*)(wr + kb + 16 + 8 * hl);
    acc = __builtin_amdgcn_wmma_f32_16x16x32_bf16(
        false, a.v, false, bfrag.v, (short)0, acc, false, false);
  }

  const float bv  = bias[n0 + ln];
  const int hidx  = (n0 + ln) / HD;
  const int didx  = (n0 + ln) % HD;
#pragma unroll
  for (int v = 0; v < 8; ++v) {
    const int row = m0 + v + 8 * hl;
    const int b   = row / SS;
    const int s   = row % SS;
    const float val = acc[v] + bv;
    if (vtrans) {
      out[((size_t)(b * HH + hidx) * HD + didx) * SS + s] = f2bf(val);
    } else {
      out[((size_t)(b * HH + hidx) * SS + s) * HD + didx] = f2bf(val);
    }
  }
}

// One workgroup = 8 waves = all 8 heads of one (b, 16-query tile).
// Gate tiles staged through LDS (async, double-buffered) so each gate byte
// crosses HBM exactly once.  Qh,Kh: [B,H,S,64] bf16 ; Vt: [B,H,64,S] bf16.
__global__ __launch_bounds__(256)
void attn16(const __bf16* __restrict__ Qh, const __bf16* __restrict__ Kh,
            const __bf16* __restrict__ Vt, const float* __restrict__ gate,
            const int* __restrict__ mask, float* __restrict__ out) {
  __shared__ float  gbuf[2 * GTILE];        // 33792 B
  __shared__ __bf16 plds[8 * 16 * 32];      // 8192 B, per-wave slices

  const int tid = threadIdx.x;
  const int h   = tid >> 5;                 // wave id == head
  const int l   = tid & 31;
  const int ln  = l & 15;
  const int hl  = l >> 4;
  const int q0  = blockIdx.x * 16;
  const int b   = blockIdx.y;
  const float L2E = 1.4426950408889634f;

  const __bf16* Qbase = Qh + (size_t)(b * HH + h) * SS * HD;
  const __bf16* Kbase = Kh + (size_t)(b * HH + h) * SS * HD;
  const __bf16* Vbase = Vt + (size_t)(b * HH + h) * HD * SS;
  const float*  Gwg   = gate + (size_t)b * SS * SS * HH + (size_t)q0 * HH;
  const int*    Mbase = mask + (size_t)b * SS * SS;
  __bf16* pw = plds + h * (16 * 32);

  // cooperative gate staging: thread -> (k-row chunk, 16-float part)
  const int chunk = tid >> 3;               // 0..31
  const int part  = tid & 7;                // 0..7
  const float* gsrc0 = Gwg + (size_t)chunk * SS * HH + part * 16;
  float* ldst0 = gbuf + chunk * GROW + part * 16;

  // Q A-fragments (persist)
  U16 a0, a1;
  {
    const __bf16* qr = Qbase + (size_t)(q0 + ln) * HD;
    a0.v = *(const v16bf*)(qr + 16 * hl);
    a1.v = *(const v16bf*)(qr + 32 + 16 * hl);
  }

  v8f o0 = {}, o1 = {}, o2 = {}, o3 = {};
  v8f mrun, lrun;
#pragma unroll
  for (int v = 0; v < 8; ++v) { mrun[v] = -3.0e38f; lrun[v] = 0.0f; }

#ifdef HAVE_ASYNC
  stage16(gsrc0, ldst0);                    // prefetch tile 0 into buf 0
#endif

  int buf = 0;
  for (int kb = 0; kb < SS; kb += 32, buf ^= 1) {
#ifdef HAVE_ASYNC
    if (kb + 32 < SS) {
      stage16(gsrc0 + (size_t)(kb + 32) * SS * HH, ldst0 + (buf ^ 1) * GTILE);
      async_wait4();                        // this step's batch complete
    } else {
      async_wait0();
    }
#else
    stage16_sync(gsrc0 + (size_t)kb * SS * HH, ldst0 + buf * GTILE);
#endif
    __syncthreads();                        // gate tile visible to all waves
    const float* gb = gbuf + buf * GTILE;

    v8f s0, s1;
#pragma unroll
    for (int t = 0; t < 2; ++t) {
      const int kt = kb + t * 16;
      U16 kf0, kf1;
      const __bf16* kr = Kbase + (size_t)(kt + ln) * HD + 16 * hl;
      kf0.v = *(const v16bf*)(kr);
      kf1.v = *(const v16bf*)(kr + 32);
      v8f c = {};
      c = __builtin_amdgcn_wmma_f32_16x16x32_bf16(
          false, a0.v, false, kf0.v, (short)0, c, false, false);
      c = __builtin_amdgcn_wmma_f32_16x16x32_bf16(
          false, a1.v, false, kf1.v, (short)0, c, false, false);
#pragma unroll
      for (int v = 0; v < 8; ++v) {
        const int m = v + 8 * hl;
        const float g = gb[(t * 16 + ln) * GROW + m * 8 + h];
        float sc = c[v] * 0.125f + g;
        const int mk = Mbase[(size_t)(q0 + m) * SS + (kt + ln)];
        if (mk == 0) sc = -1.0e9f;
        c[v] = sc;
      }
      if (t == 0) s0 = c; else s1 = c;
    }

    // online softmax (rows live across 16 lanes of each half-wave)
    v8f mnew, alpha;
#pragma unroll
    for (int v = 0; v < 8; ++v) {
      float t = fmaxf(s0[v], s1[v]);
#pragma unroll
      for (int off = 1; off < 16; off <<= 1) t = fmaxf(t, __shfl_xor(t, off, 32));
      mnew[v] = fmaxf(mrun[v], t);
    }
#pragma unroll
    for (int v = 0; v < 8; ++v) {
      alpha[v] = __builtin_amdgcn_exp2f((mrun[v] - mnew[v]) * L2E);
      float p0 = __builtin_amdgcn_exp2f((s0[v] - mnew[v]) * L2E);
      float p1 = __builtin_amdgcn_exp2f((s1[v] - mnew[v]) * L2E);
      s0[v] = p0; s1[v] = p1;
      float rs = p0 + p1;
#pragma unroll
      for (int off = 1; off < 16; off <<= 1) rs += __shfl_xor(rs, off, 32);
      lrun[v] = lrun[v] * alpha[v] + rs;
      o0[v] *= alpha[v]; o1[v] *= alpha[v];
      o2[v] *= alpha[v]; o3[v] *= alpha[v];
    }
    mrun = mnew;

    // reshape P (C-layout -> A-layout) via per-wave LDS slice
#pragma unroll
    for (int v = 0; v < 8; ++v) {
      pw[(v + 8 * hl) * 32 + ln]      = f2bf(s0[v]);
      pw[(v + 8 * hl) * 32 + 16 + ln] = f2bf(s1[v]);
    }
    lds_wait0();                             // wave-local RAW on LDS
    U16 pa;
    pa.v = *(const v16bf*)(pw + ln * 32 + 16 * hl);

    // O += P * V
    const __bf16* vr = Vbase + (size_t)ln * SS + kb + 16 * hl;
    U16 vb0, vb1, vb2, vb3;
    vb0.v = *(const v16bf*)(vr);
    vb1.v = *(const v16bf*)(vr + (size_t)16 * SS);
    vb2.v = *(const v16bf*)(vr + (size_t)32 * SS);
    vb3.v = *(const v16bf*)(vr + (size_t)48 * SS);
    o0 = __builtin_amdgcn_wmma_f32_16x16x32_bf16(false, pa.v, false, vb0.v, (short)0, o0, false, false);
    o1 = __builtin_amdgcn_wmma_f32_16x16x32_bf16(false, pa.v, false, vb1.v, (short)0, o1, false, false);
    o2 = __builtin_amdgcn_wmma_f32_16x16x32_bf16(false, pa.v, false, vb2.v, (short)0, o2, false, false);
    o3 = __builtin_amdgcn_wmma_f32_16x16x32_bf16(false, pa.v, false, vb3.v, (short)0, o3, false, false);

    __syncthreads();                         // all waves done with gbuf[buf]
  }

  // epilogue: normalize and store f32 output [B,S,512]
#pragma unroll
  for (int v = 0; v < 8; ++v) {
    const int m = v + 8 * hl;
    const float inv = 1.0f / lrun[v];
    float* orow = out + ((size_t)b * SS + (q0 + m)) * DDIM + h * HD;
    orow[ln]      = o0[v] * inv;
    orow[16 + ln] = o1[v] * inv;
    orow[32 + ln] = o2[v] * inv;
    orow[48 + ln] = o3[v] * inv;
  }
}

extern "C" void kernel_launch(void* const* d_in, const int* in_sizes, int n_in,
                              void* d_out, int out_size, void* d_ws, size_t ws_size,
                              hipStream_t stream) {
  const float* queries = (const float*)d_in[0];
  const float* keys    = (const float*)d_in[1];
  const float* values  = (const float*)d_in[2];
  const float* gate    = (const float*)d_in[3];
  const int*   maskp   = (const int*)d_in[4];
  const float* Wq      = (const float*)d_in[5];
  const float* bq      = (const float*)d_in[6];
  const float* Wk      = (const float*)d_in[7];
  const float* bk      = (const float*)d_in[8];
  const float* Wv      = (const float*)d_in[9];
  const float* bv      = (const float*)d_in[10];
  float* out = (float*)d_out;

  const size_t nX = (size_t)BB * SS * DDIM;
  const size_t nW = (size_t)DDIM * DDIM;

  char* w = (char*)d_ws;
  __bf16* Xq  = (__bf16*)w; w += nX * 2;
  __bf16* Xk  = (__bf16*)w; w += nX * 2;
  __bf16* Xv  = (__bf16*)w; w += nX * 2;
  __bf16* Wqb = (__bf16*)w; w += nW * 2;
  __bf16* Wkb = (__bf16*)w; w += nW * 2;
  __bf16* Wvb = (__bf16*)w; w += nW * 2;
  __bf16* Qh  = (__bf16*)w; w += nX * 2;      // [B,H,S,64]
  __bf16* Kh  = (__bf16*)w; w += nX * 2;      // [B,H,S,64]
  __bf16* Vt  = (__bf16*)w; w += nX * 2;      // [B,H,64,S]

  const int cb = 256;
  cvt_f32_to_bf16<<<(unsigned)((nX + cb - 1) / cb), cb, 0, stream>>>(queries, Xq, (int)nX);
  cvt_f32_to_bf16<<<(unsigned)((nX + cb - 1) / cb), cb, 0, stream>>>(keys,    Xk, (int)nX);
  cvt_f32_to_bf16<<<(unsigned)((nX + cb - 1) / cb), cb, 0, stream>>>(values,  Xv, (int)nX);
  cvt_f32_to_bf16<<<(unsigned)((nW + cb - 1) / cb), cb, 0, stream>>>(Wq, Wqb, (int)nW);
  cvt_f32_to_bf16<<<(unsigned)((nW + cb - 1) / cb), cb, 0, stream>>>(Wk, Wkb, (int)nW);
  cvt_f32_to_bf16<<<(unsigned)((nW + cb - 1) / cb), cb, 0, stream>>>(Wv, Wvb, (int)nW);

  dim3 pg((BB * SS) / 16, DDIM / 16);
  proj16x16<<<pg, 32, 0, stream>>>(Xq, Wqb, bq, Qh, 0);
  proj16x16<<<pg, 32, 0, stream>>>(Xk, Wkb, bk, Kh, 0);
  proj16x16<<<pg, 32, 0, stream>>>(Xv, Wvb, bv, Vt, 1);

  dim3 ag(SS / 16, BB);                 // 128 x 2 workgroups, 8 waves each
  attn16<<<ag, 256, 0, stream>>>(Qh, Kh, Vt, gate, maskp, out);
}
